// HadLinear_24215025615050
// MI455X (gfx1250) — matrix operand order repaired and checked by reading, not compiled, not run on hardware
//
#include <hip/hip_runtime.h>

// ---------------------------------------------------------------------------
// HadLinear for MI455X (gfx1250, wave32, WMMA)
//   out = (FWHT_1024(x)/32) @ W^T  ==  x @ (W*H/32)^T     (H symmetric)
// Plan:
//   K1: W' = bf16( FWHT_1024(W rows) / 32 )     (4096 rows, 4 blocks each)
//   K2: Xb = bf16(x)                            (16384 x 4096)
//   K3: out = Xb @ W'^T  via v_wmma_f32_16x16x32_bf16, fp32 accumulate,
//       double-buffered LDS tiles filled with global_load_async_to_lds_b128
//       (ASYNCcnt-tracked, overlapped with WMMA compute).
// Working set Xb+W' = 160 MB < 192 MB L2 -> GEMM K-loop streams from L2.
// ---------------------------------------------------------------------------

typedef __attribute__((ext_vector_type(16))) __bf16 v16bf;
typedef __attribute__((ext_vector_type(8)))  float  v8f;

#define D_IN   4096
#define D_OUT  4096
#define HADB   1024
#define BM     128
#define BN     128
#define BK     64
#define NT     (D_IN / BK)          // 64 K-tiles

static __device__ __forceinline__ unsigned short f32_to_bf16(float f) {
    union { float f; unsigned int u; } c; c.f = f;
    unsigned int u = c.u;
    u += 0x7FFFu + ((u >> 16) & 1u);          // round-to-nearest-even
    return (unsigned short)(u >> 16);
}
static __device__ __forceinline__ unsigned int pack2_bf16(float a, float b) {
    return (unsigned int)f32_to_bf16(a) | ((unsigned int)f32_to_bf16(b) << 16);
}

// Generic shared pointer -> 32-bit LDS byte address (low half of aperture form)
static __device__ __forceinline__ unsigned lds_addr_of(const void* p) {
    return (unsigned)(unsigned long long)p;
}

// 64 contiguous bytes: 4x async b128, memory -> LDS.  INST_OFFSET is added to
// both the global and LDS addresses, so one (lds,voff) pair serves all 4.
static __device__ __forceinline__ void async_copy64(
    const unsigned short* __restrict__ gbase, unsigned goff_bytes, unsigned lds_bytes) {
    asm volatile(
        "global_load_async_to_lds_b128 %0, %1, %2 offset:0\n\t"
        "global_load_async_to_lds_b128 %0, %1, %2 offset:16\n\t"
        "global_load_async_to_lds_b128 %0, %1, %2 offset:32\n\t"
        "global_load_async_to_lds_b128 %0, %1, %2 offset:48"
        :
        : "v"(lds_bytes), "v"(goff_bytes), "s"(gbase)
        : "memory");
}

// --- Kernel 1: per-1024-block FWHT of one weight row chunk, scale, -> bf16 ---
__global__ __launch_bounds__(256)
void fwht_weight_kernel(const float* __restrict__ w, unsigned short* __restrict__ wp) {
    __shared__ float s[HADB];
    const int blk = blockIdx.x;            // 0 .. D_OUT*(D_IN/HADB)-1
    const int row = blk >> 2;              // weight output-feature row
    const int cb  = blk & 3;               // which 1024-block of the row
    const int t   = threadIdx.x;           // 256 threads, 4 elems each

    const float4 v = ((const float4*)(w + (size_t)row * D_IN + cb * HADB))[t];
    s[4*t+0] = v.x; s[4*t+1] = v.y; s[4*t+2] = v.z; s[4*t+3] = v.w;
    __syncthreads();

    for (int h = 1; h < HADB; h <<= 1) {
        #pragma unroll
        for (int r = 0; r < 2; ++r) {      // 512 butterflies / 256 threads
            const int i   = t + r * 256;
            const int idx = ((i & ~(h - 1)) << 1) | (i & (h - 1));
            const float a = s[idx], b = s[idx + h];
            s[idx]     = a + b;
            s[idx + h] = a - b;
        }
        __syncthreads();
    }

    const float sc = 0.03125f;             // 1/sqrt(1024)
    uint2 p;
    p.x = pack2_bf16(s[4*t+0] * sc, s[4*t+1] * sc);
    p.y = pack2_bf16(s[4*t+2] * sc, s[4*t+3] * sc);
    ((uint2*)(wp + (size_t)row * D_IN + cb * HADB))[t] = p;
}

// --- Kernel 2: x fp32 -> bf16 (vectorized, 4 elems/thread) ------------------
__global__ __launch_bounds__(256)
void convert_x_kernel(const float* __restrict__ x, unsigned short* __restrict__ xb,
                      long n4) {
    const long i = (long)blockIdx.x * blockDim.x + threadIdx.x;
    if (i < n4) {
        const float4 v = ((const float4*)x)[i];
        uint2 p;
        p.x = pack2_bf16(v.x, v.y);
        p.y = pack2_bf16(v.z, v.w);
        ((uint2*)xb)[i] = p;
    }
}

// --- Kernel 3: out[M,4096] = Xb[M,4096] @ W'[4096,4096]^T  (bf16 WMMA) ------
__global__ __launch_bounds__(256)
void had_gemm_kernel(const unsigned short* __restrict__ Xb,
                     const unsigned short* __restrict__ Wb,
                     float* __restrict__ out, int Mtotal) {
    // double-buffered tiles: A[2][128*64] + B[2][128*64] bf16 = 64 KB
    __shared__ __align__(16) unsigned short smem[4 * BM * BK];
    unsigned short* sA = smem;                 // [2][BM*BK]
    unsigned short* sB = smem + 2 * BM * BK;   // [2][BN*BK]

    const int tid  = threadIdx.x;
    const int lane = tid & 31;
    const int wave = tid >> 5;               // 8 waves
    const int wm   = wave >> 2;              // 0..1  (M)
    const int wn   = wave & 3;               // 0..3  (N)
    const int m0   = blockIdx.y * BM;
    const int n0   = blockIdx.x * BN;

    // per-thread async-copy mapping: row r (2 threads/row), 64B half each
    const int r    = tid >> 1;               // 0..127
    const int half = tid & 1;                // 0..1
    const unsigned goffA0 = (unsigned)(((m0 + r) * D_IN) * 2 + half * 64);
    const unsigned goffB0 = (unsigned)(((n0 + r) * D_IN) * 2 + half * 64);
    const unsigned ldsA0  = lds_addr_of(sA) + (unsigned)(r * BK * 2 + half * 64);
    const unsigned ldsB0  = lds_addr_of(sB) + (unsigned)(r * BK * 2 + half * 64);

    v8f acc[4][2];
    #pragma unroll
    for (int mi = 0; mi < 4; ++mi)
        #pragma unroll
        for (int ni = 0; ni < 2; ++ni)
            acc[mi][ni] = (v8f){0.f,0.f,0.f,0.f,0.f,0.f,0.f,0.f};

    // prologue: tile 0 -> buffer 0
    async_copy64(Xb, goffA0, ldsA0);
    async_copy64(Wb, goffB0, ldsB0);

    for (int it = 0; it < NT; ++it) {
        const int p = it & 1;

        if (it + 1 < NT) {
            // issue tile it+1 into the other buffer; overlaps this tile's WMMAs
            const unsigned kb = (unsigned)((it + 1) * BK * 2);
            const unsigned bb = (unsigned)((1 - p) * BM * BK * 2);
            async_copy64(Xb, goffA0 + kb, ldsA0 + bb);
            async_copy64(Wb, goffB0 + kb, ldsB0 + bb);
            asm volatile("s_wait_asynccnt 0x8" ::: "memory");  // tile it done
        } else {
            asm volatile("s_wait_asynccnt 0x0" ::: "memory");
        }
        __syncthreads();

        const unsigned short* tA = sA + p * (BM * BK);
        const unsigned short* tB = sB + p * (BN * BK);

        #pragma unroll
        for (int k32 = 0; k32 < BK; k32 += 32) {
            // A fragments (ISA 16-bit A 16x32 layout): lane row = lane&15,
            // two 16B chunks at K offsets (lane>>4)*8 and 16+(lane>>4)*8
            const int arow = lane & 15;
            const int asel = (lane >> 4) * 8;
            v16bf af[4];
            #pragma unroll
            for (int mi = 0; mi < 4; ++mi) {
                const unsigned short* pa =
                    tA + (wm * 64 + mi * 16 + arow) * BK + k32 + asel;
                union { v16bf v; int4 q[2]; } u;
                u.q[0] = *(const int4*)(pa);
                u.q[1] = *(const int4*)(pa + 16);
                af[mi] = u.v;
            }
            // B fragments (32x16): lane col = lane&15 (a W' row),
            // two 16B chunks at K offsets (lane>>4)*8 and 16+(lane>>4)*8
            const int bsel = (lane >> 4) * 8;
            v16bf bfr[2];
            #pragma unroll
            for (int ni = 0; ni < 2; ++ni) {
                const unsigned short* pb =
                    tB + (wn * 32 + ni * 16 + (lane & 15)) * BK + k32 + bsel;
                union { v16bf v; int4 q[2]; } u;
                u.q[0] = *(const int4*)(pb);
                u.q[1] = *(const int4*)(pb + 16);
                bfr[ni] = u.v;
            }
            #pragma unroll
            for (int mi = 0; mi < 4; ++mi)
                #pragma unroll
                for (int ni = 0; ni < 2; ++ni)
                    acc[mi][ni] = __builtin_amdgcn_wmma_f32_16x16x32_bf16(
                        false, af[mi], false, bfr[ni],
                        (short)0, acc[mi][ni], false, false);
        }
        __syncthreads();   // all waves done reading buf p before it is refilled
    }

    // ---- epilogue: C layout = 8 VGPRs, lane<16 -> M=j, lane>=16 -> M=j+8 ----
    #pragma unroll
    for (int mi = 0; mi < 4; ++mi)
        #pragma unroll
        for (int ni = 0; ni < 2; ++ni)
            #pragma unroll
            for (int j = 0; j < 8; ++j) {
                const int m = m0 + wm * 64 + mi * 16 + j + ((lane >> 4) * 8);
                const int n = n0 + wn * 32 + ni * 16 + (lane & 15);
                out[(size_t)m * D_OUT + n] = acc[mi][ni][j];
            }
    (void)Mtotal;
}

// ---------------------------------------------------------------------------
extern "C" void kernel_launch(void* const* d_in, const int* in_sizes, int n_in,
                              void* d_out, int out_size, void* d_ws, size_t ws_size,
                              hipStream_t stream) {
    const float* x = (const float*)d_in[0];   // (B,S,4096) fp32
    const float* w = (const float*)d_in[1];   // (4096,4096) fp32
    float* out = (float*)d_out;

    const int M = in_sizes[0] / D_IN;         // 16384 rows

    unsigned short* Wb = (unsigned short*)d_ws;                 // 32 MB
    unsigned short* Xb = Wb + (size_t)D_OUT * D_IN;             // 128 MB

    // K1: Hadamard-transform weight rows (per 1024-block), scale, -> bf16
    fwht_weight_kernel<<<dim3(D_OUT * (D_IN / HADB)), dim3(256), 0, stream>>>(w, Wb);

    // K2: x -> bf16
    const long n4 = (long)M * D_IN / 4;
    convert_x_kernel<<<dim3((unsigned)((n4 + 255) / 256)), dim3(256), 0, stream>>>(x, Xb, n4);

    // K3: WMMA GEMM (async double-buffered)
    dim3 grid(D_OUT / BN, M / BM);
    had_gemm_kernel<<<grid, dim3(256), 0, stream>>>(Xb, Wb, out, M);
}
